// ODERNNEncoder_79276506349918
// MI455X (gfx1250) — matrix operand order repaired and checked by reading, not compiled, not run on hardware
//
#include <hip/hip_runtime.h>
#include <hip/hip_bf16.h>

// ODE-RNN encoder, MI455X (gfx1250, wave32, WMMA).
// B=4096,T=100,D=64,H=128. Per-workgroup: 64 batch rows resident for all T
// steps (h state in LDS, no barriers in main loop: each wave owns a private
// 16-row slab and per-wave LDS ops are in-order). All GEMMs via
// v_wmma_f32_16x16x32_bf16; tanh/sigmoid via hardware v_tanh_f32 (batched
// 8-wide). B fragments are explicitly preloaded into distinct register
// arrays per gate/tile so loads clause together and WMMA chains don't eat
// one L2/LDS round-trip per instruction.

#define B_  4096
#define T_  100
#define D_  64
#define H_  128
#define BT  64      // batch rows per workgroup (4 waves x 16 rows)
#define PW  136     // bf16 LDS pitch (272B rows -> 4-bank rotation, conflict-free b128)
#define PF  132     // f32  LDS pitch (528B rows -> conflict-free b32 column reads)

typedef __attribute__((ext_vector_type(16))) __bf16 v16bf;
typedef __attribute__((ext_vector_type(8)))  __bf16 v8bf;
typedef __attribute__((ext_vector_type(8)))  float  v8f;

struct __align__(16) Smem {
  __bf16 w1[128 * PW];     // W1 bf16, row-major (row n = output col n)
  __bf16 w2[128 * PW];     // W2 bf16
  __bf16 a[BT * PW];       // bf16 A staging: h / htmp (RK4 stage input)
  __bf16 u[BT * PW];       // bf16 staging: tanh() output u, and x tile for GRU
  float  h[BT * PF];       // f32 hidden state, resident across all T steps
  float  b1v[128], b2v[128];
  float  bih[384], bhh[384];
};

static __device__ __forceinline__ v8f vzero() {
  v8f z;
#pragma unroll
  for (int i = 0; i < 8; ++i) z[i] = 0.f;
  return z;
}

// 8x hardware tanh (CDNA5 TRANS op). Consecutive v_tanh ops are mutually
// independent, satisfying the "1 independent op after TRANS" rule for all but
// the last, which gets the single v_nop.
static __device__ __forceinline__ v8f htanh8(v8f x) {
  float x0 = x[0], x1 = x[1], x2 = x[2], x3 = x[3];
  float x4 = x[4], x5 = x[5], x6 = x[6], x7 = x[7];
  asm("v_tanh_f32 %0, %0\n\t"
      "v_tanh_f32 %1, %1\n\t"
      "v_tanh_f32 %2, %2\n\t"
      "v_tanh_f32 %3, %3\n\t"
      "v_tanh_f32 %4, %4\n\t"
      "v_tanh_f32 %5, %5\n\t"
      "v_tanh_f32 %6, %6\n\t"
      "v_tanh_f32 %7, %7\n\t"
      "v_nop"
      : "+v"(x0), "+v"(x1), "+v"(x2), "+v"(x3),
        "+v"(x4), "+v"(x5), "+v"(x6), "+v"(x7));
  v8f r;
  r[0] = x0; r[1] = x1; r[2] = x2; r[3] = x3;
  r[4] = x4; r[5] = x5; r[6] = x6; r[7] = x7;
  return r;
}
// sigmoid(x) = 0.5 + 0.5*tanh(x/2), vectorized.
static __device__ __forceinline__ v8f sigm8(v8f x) {
  return 0.5f * htanh8(0.5f * x) + 0.5f;
}

// A fragment (16x32 bf16): lanes 0-15 row=lane K{kb..kb+15},
// lanes 16-31 same rows, K shifted by 16.  slab = 16-row LDS slab, pitch PW.
static __device__ __forceinline__ v16bf load_a_frag(const __bf16* slab, int lane, int kc) {
  int r  = lane & 15;
  int kb = kc * 32 + ((lane & 16) ? 16 : 0);
  const __bf16* p = slab + r * PW + kb;
  v8bf lo = *(const v8bf*)(p);
  v8bf hi = *(const v8bf*)(p + 8);
  return __builtin_shufflevector(lo, hi, 0, 1, 2, 3, 4, 5, 6, 7,
                                 8, 9, 10, 11, 12, 13, 14, 15);
}

// B fragment (32x16 bf16): lane n holds column n of B = row n of W (since
// B[k][n] = W[n][k] for h @ W^T). 32 contiguous bytes per lane.
static __device__ __forceinline__ v16bf load_b_lds(const __bf16* w, int lane, int nt, int kc) {
  int n  = nt * 16 + (lane & 15);
  int kb = kc * 32 + ((lane & 16) ? 16 : 0);
  const __bf16* p = w + n * PW + kb;
  v8bf lo = *(const v8bf*)(p);
  v8bf hi = *(const v8bf*)(p + 8);
  return __builtin_shufflevector(lo, hi, 0, 1, 2, 3, 4, 5, 6, 7,
                                 8, 9, 10, 11, 12, 13, 14, 15);
}

static __device__ __forceinline__ v16bf load_b_g(const __bf16* w, int K, int lane, int nt, int kc) {
  int n  = nt * 16 + (lane & 15);
  int kb = kc * 32 + ((lane & 16) ? 16 : 0);
  const __bf16* p = w + n * K + kb;
  v8bf lo = *(const v8bf*)(p);
  v8bf hi = *(const v8bf*)(p + 8);
  return __builtin_shufflevector(lo, hi, 0, 1, 2, 3, 4, 5, 6, 7,
                                 8, 9, 10, 11, 12, 13, 14, 15);
}

static __device__ __forceinline__ v8f wmma_bf(v16bf a, v16bf b, v8f c) {
  return __builtin_amdgcn_wmma_f32_16x16x32_bf16(false, a, false, b, (short)0, c, false, false);
}

__global__ void cvt_bf16_kernel(const float* __restrict__ src, __bf16* __restrict__ dst, int n) {
  int i = blockIdx.x * 256 + threadIdx.x;
  if (i < n) dst[i] = (__bf16)src[i];
}

__global__ void __launch_bounds__(128)
__attribute__((amdgpu_num_vgpr(256)))
odernn_kernel(const float* __restrict__ x, const float* __restrict__ tq,
              const float* __restrict__ b1g, const float* __restrict__ b2g,
              const float* __restrict__ bihg, const float* __restrict__ bhhg,
              const float* __restrict__ bmug, const float* __restrict__ blvg,
              const __bf16* __restrict__ ws, float* __restrict__ out) {
  extern __shared__ char smraw[];
  Smem& sm = *(Smem*)smraw;
  const int tid  = threadIdx.x;
  const int wave = tid >> 5;
  const int lane = tid & 31;
  const int bbase = blockIdx.x * BT;

  const __bf16* wsW1  = ws;
  const __bf16* wsW2  = ws + 16384;
  const __bf16* wsWhh = ws + 32768;   // [384][128]
  const __bf16* wsWih = ws + 81920;   // [384][64]
  const __bf16* wsWmu = ws + 106496;  // [128][128]
  const __bf16* wsWlv = ws + 122880;  // [128][128]

  // ---- one-time init: weights -> LDS, biases, h = 0 ----
  for (int i = tid; i < 128 * 128; i += 128) {
    int r = i >> 7, c = i & 127;
    sm.w1[r * PW + c] = wsW1[i];
    sm.w2[r * PW + c] = wsW2[i];
  }
  if (tid < 128) { sm.b1v[tid] = b1g[tid]; sm.b2v[tid] = b2g[tid]; }
  for (int i = tid; i < 384; i += 128) { sm.bih[i] = bihg[i]; sm.bhh[i] = bhhg[i]; }
  for (int i = tid; i < BT * PW; i += 128) sm.a[i] = (__bf16)0.f;
  for (int i = tid; i < BT * PF; i += 128) sm.h[i] = 0.f;
  __syncthreads();  // only barrier in the kernel

  __bf16* aslab = sm.a + wave * 16 * PW;  // this wave's 16-row bf16 h slab
  __bf16* uslab = sm.u + wave * 16 * PW;  // this wave's staging slab
  const int col_l = lane & 15;            // C-layout: col within tile
  const int roff  = (lane & 16) ? 8 : 0;  // C-layout: row offset within tile
  const int rloc0 = wave * 16 + roff;     // LDS row base for C rows

  float tprev = 0.f;
#pragma unroll 1
  for (int t = 0; t < T_; ++t) {
    float tcur = tq[t];                   // t_seq[0][t] (uniform scalar load)
    float dt   = (t == 0) ? 0.f : (tcur - tprev);
    tprev = tcur;
    const float sub  = dt * 0.25f;        // N_RK4 = 4
    const float sub6 = sub * (1.f / 6.f);

    // ================= RK4 jump: 4 substeps fused (dt=0 -> identity) ======
    v8f ks[8];
#pragma unroll 1
    for (int s = 0; s < 4; ++s) {
      // ---- GEMM1: u = tanh(htmp @ W1^T + b1) -> uslab ----
      v16bf af[4];
#pragma unroll
      for (int kc = 0; kc < 4; ++kc) af[kc] = load_a_frag(aslab, lane, kc);
#pragma unroll
      for (int nt = 0; nt < 8; ++nt) {
        v16bf bw[4];
#pragma unroll
        for (int kc = 0; kc < 4; ++kc) bw[kc] = load_b_lds(sm.w1, lane, nt, kc);
        v8f acc = vzero();
#pragma unroll
        for (int kc = 0; kc < 4; ++kc) acc = wmma_bf(af[kc], bw[kc], acc);
        int col = nt * 16 + col_l;
        acc = htanh8(acc + sm.b1v[col]);
#pragma unroll
        for (int v = 0; v < 8; ++v)
          uslab[(roff + v) * PW + col] = (__bf16)acc[v];
      }
      // ---- GEMM2: k = u @ W2^T + b2; RK4 accumulate + next stage input ----
#pragma unroll
      for (int kc = 0; kc < 4; ++kc) af[kc] = load_a_frag(uslab, lane, kc);
#pragma unroll
      for (int nt = 0; nt < 8; ++nt) {
        v16bf bw[4];
#pragma unroll
        for (int kc = 0; kc < 4; ++kc) bw[kc] = load_b_lds(sm.w2, lane, nt, kc);
        v8f acc = vzero();
#pragma unroll
        for (int kc = 0; kc < 4; ++kc) acc = wmma_bf(af[kc], bw[kc], acc);
        int col = nt * 16 + col_l;
        acc += sm.b2v[col];
        if (s == 0) ks[nt] = acc;
        else        ks[nt] += ((s == 3) ? 1.f : 2.f) * acc;
        if (s < 3) {
          float c = (s == 2) ? sub : 0.5f * sub;
#pragma unroll
          for (int v = 0; v < 8; ++v) {
            float hv = sm.h[(rloc0 + v) * PF + col];
            aslab[(roff + v) * PW + col] = (__bf16)(hv + c * acc[v]);
          }
        } else {
#pragma unroll
          for (int v = 0; v < 8; ++v) {
            float hv = sm.h[(rloc0 + v) * PF + col] + sub6 * ks[nt][v];
            sm.h[(rloc0 + v) * PF + col] = hv;
            aslab[(roff + v) * PW + col] = (__bf16)hv;
          }
        }
      }
    }

    // ================= GRU cell ===========================================
    // stage x[rows, t, :] -> uslab (bf16, cols 0..63) + prefetch next step
    for (int e = lane; e < 512; e += 32) {
      int r  = e >> 5;
      int c2 = (e & 31) * 2;
      size_t gb = (size_t)(bbase + wave * 16 + r);
      const float* xp = x + gb * (size_t)(T_ * D_) + (size_t)t * D_ + c2;
      float2 xv = *(const float2*)xp;
      uslab[r * PW + c2]     = (__bf16)xv.x;
      uslab[r * PW + c2 + 1] = (__bf16)xv.y;
      if (c2 == 0 && t + 1 < T_) __builtin_prefetch(xp + D_, 0, 0);
    }
    v16bf ax[2];
#pragma unroll
    for (int kc = 0; kc < 2; ++kc) ax[kc] = load_a_frag(uslab, lane, kc);
    v16bf ah[4];
#pragma unroll
    for (int kc = 0; kc < 4; ++kc) ah[kc] = load_a_frag(aslab, lane, kc);

    // Per output tile: all four gate GEMM slices + elementwise combine.
    // Tiles independent; per-gate B fragments preloaded into distinct arrays
    // and staggered one gate ahead so L2 latency overlaps the WMMA chains.
#pragma unroll
    for (int nt = 0; nt < 8; ++nt) {
      int col = nt * 16 + col_l;
      // preload hn-gate and r-gate fragments up front
      v16bf bhn[4];
#pragma unroll
      for (int kc = 0; kc < 4; ++kc)
        bhn[kc] = load_b_g(wsWhh, 128, lane, 16 + nt, kc);
      v16bf br[6];
#pragma unroll
      for (int kc = 0; kc < 2; ++kc)
        br[kc] = load_b_g(wsWih, 64, lane, nt, kc);
#pragma unroll
      for (int kc = 0; kc < 4; ++kc)
        br[2 + kc] = load_b_g(wsWhh, 128, lane, nt, kc);
      // h_n = h @ Whh[256:384]^T + bhh_n
      v8f hn = vzero();
#pragma unroll
      for (int kc = 0; kc < 4; ++kc) hn = wmma_bf(ah[kc], bhn[kc], hn);
      hn += sm.bhh[256 + col];
      // preload n-gate fragments while r chain runs
      v16bf bn[2];
#pragma unroll
      for (int kc = 0; kc < 2; ++kc)
        bn[kc] = load_b_g(wsWih, 64, lane, 16 + nt, kc);
      // r = sigmoid(x@Wih_r + h@Whh_r + b); hn <- r * h_n
      v8f ar = vzero();
#pragma unroll
      for (int kc = 0; kc < 2; ++kc) ar = wmma_bf(ax[kc], br[kc], ar);
#pragma unroll
      for (int kc = 0; kc < 4; ++kc) ar = wmma_bf(ah[kc], br[2 + kc], ar);
      ar += (sm.bih[col] + sm.bhh[col]);
      hn *= sigm8(ar);
      // preload z-gate fragments while n chain runs
      v16bf bz[6];
#pragma unroll
      for (int kc = 0; kc < 2; ++kc)
        bz[kc] = load_b_g(wsWih, 64, lane, 8 + nt, kc);
#pragma unroll
      for (int kc = 0; kc < 4; ++kc)
        bz[2 + kc] = load_b_g(wsWhh, 128, lane, 8 + nt, kc);
      // n = tanh(x@Wih_n + b_ih_n + r*h_n)
      v8f an = vzero();
#pragma unroll
      for (int kc = 0; kc < 2; ++kc) an = wmma_bf(ax[kc], bn[kc], an);
      an += sm.bih[256 + col];
      an = htanh8(an + hn);
      // z = sigmoid(x@Wih_z + h@Whh_z + b)
      v8f az = vzero();
#pragma unroll
      for (int kc = 0; kc < 2; ++kc) az = wmma_bf(ax[kc], bz[kc], az);
#pragma unroll
      for (int kc = 0; kc < 4; ++kc) az = wmma_bf(ah[kc], bz[2 + kc], az);
      az += (sm.bih[128 + col] + sm.bhh[128 + col]);
      v8f z = sigm8(az);
      // h = (1-z)*n + z*h
#pragma unroll
      for (int v = 0; v < 8; ++v) {
        float hv = sm.h[(rloc0 + v) * PF + col];
        float hnew = (1.f - z[v]) * an[v] + z[v] * hv;
        sm.h[(rloc0 + v) * PF + col] = hnew;
        aslab[(roff + v) * PW + col] = (__bf16)hnew;
      }
    }
  }

  // ================= output projections: mu, logvar =======================
  v16bf ah[4];
#pragma unroll
  for (int kc = 0; kc < 4; ++kc) ah[kc] = load_a_frag(aslab, lane, kc);
#pragma unroll
  for (int o = 0; o < 2; ++o) {
    const __bf16* wsel = o ? wsWlv : wsWmu;
    const float*  bsel = o ? blvg : bmug;
    float* obase = out + (size_t)o * B_ * H_;
#pragma unroll
    for (int nt = 0; nt < 8; ++nt) {
      v16bf bw[4];
#pragma unroll
      for (int kc = 0; kc < 4; ++kc)
        bw[kc] = load_b_g(wsel, 128, lane, nt, kc);
      v8f acc = vzero();
#pragma unroll
      for (int kc = 0; kc < 4; ++kc) acc = wmma_bf(ah[kc], bw[kc], acc);
      int col = nt * 16 + col_l;
      float bb = bsel[col];
#pragma unroll
      for (int v = 0; v < 8; ++v)
        obase[(size_t)(bbase + rloc0 + v) * H_ + col] = acc[v] + bb;
    }
  }
}

extern "C" void kernel_launch(void* const* d_in, const int* in_sizes, int n_in,
                              void* d_out, int out_size, void* d_ws, size_t ws_size,
                              hipStream_t stream) {
  const float* x   = (const float*)d_in[0];
  const float* tq  = (const float*)d_in[1];   // t_seq; row 0 used: tq[t]
  const float* W1  = (const float*)d_in[2];
  const float* b1  = (const float*)d_in[3];
  const float* W2  = (const float*)d_in[4];
  const float* b2  = (const float*)d_in[5];
  const float* Wih = (const float*)d_in[6];
  const float* Whh = (const float*)d_in[7];
  const float* bih = (const float*)d_in[8];
  const float* bhh = (const float*)d_in[9];
  const float* Wmu = (const float*)d_in[10];
  const float* bmu = (const float*)d_in[11];
  const float* Wlv = (const float*)d_in[12];
  const float* blv = (const float*)d_in[13];
  __bf16* ws = (__bf16*)d_ws;
  float* out = (float*)d_out;

  // one-time per-call weight conversion f32 -> bf16 into workspace
  cvt_bf16_kernel<<<64, 256, 0, stream>>>(W1, ws + 0, 16384);
  cvt_bf16_kernel<<<64, 256, 0, stream>>>(W2, ws + 16384, 16384);
  cvt_bf16_kernel<<<192, 256, 0, stream>>>(Whh, ws + 32768, 49152);
  cvt_bf16_kernel<<<96, 256, 0, stream>>>(Wih, ws + 81920, 24576);
  cvt_bf16_kernel<<<64, 256, 0, stream>>>(Wmu, ws + 106496, 16384);
  cvt_bf16_kernel<<<64, 256, 0, stream>>>(Wlv, ws + 122880, 16384);

  // 64 workgroups x 128 threads (4 waves); 142,336 B dynamic LDS each
  odernn_kernel<<<B_ / BT, 128, sizeof(Smem), stream>>>(
      x, tq, b1, b2, bih, bhh, bmu, blv, ws, out);
}
